// GraphCAMHead_55473797595651
// MI455X (gfx1250) — matrix operand order repaired and checked by reading, not compiled, not run on hardware
//
#include <hip/hip_runtime.h>

// ---------------------------------------------------------------------------
// GraphCAM head for MI455X (gfx1250): two M=147456,N=64,K=512 GEMMs done with
// v_wmma_f32_16x16x32_bf16, fused epilogues for gumbel-argmax/heatmap and
// max-softmax cluster loss. bf16 inputs / f32 accumulate keeps the kernel at
// the HBM roofline (~57 FLOP/B vs 23.3 TB/s).
// ---------------------------------------------------------------------------

typedef __attribute__((ext_vector_type(16))) __bf16 v16bf;
typedef __attribute__((ext_vector_type(8)))  float  v8f;

#define HW   9216      // 96*96
#define CC   512
#define NN   64
#define BB   16
#define TSTR 69        // padded LDS tile stride (gcd(69,64)=1 -> conflict free)

#define OFF_LOGITS 147456
#define OFF_LOSS   147472
#define OFF_IP     147473

union AFrag { v16bf v; unsigned u[8]; };
union BFrag { uint4 q[2]; v16bf v; };

__device__ __forceinline__ unsigned short f2bf(float f) {
    unsigned u = __float_as_uint(f);
    u += 0x7FFFu + ((u >> 16) & 1u);     // round-to-nearest-even
    return (unsigned short)(u >> 16);
}

// ---------------------------------------------------------------------------
// kA: mixprop step h1 = alpha*centers + (1-alpha)*rownorm(A+I)@centers
//     (blocks 0..63) and the per-center classifier table s[n]=W_bc.centers[n]+b
//     (block 64).
// ---------------------------------------------------------------------------
__global__ void graphcam_kA(const float* __restrict__ centers,
                            const float* __restrict__ Aadj,
                            const float* __restrict__ W_bc,
                            const float* __restrict__ b_bc,
                            float* __restrict__ h1,
                            float* __restrict__ stab) {
    const int v = blockIdx.x;
    const int tid = threadIdx.x;
    if (v < NN) {
        float rs = 1.0f;                       // +I diagonal
        for (int w = 0; w < NN; ++w) rs += Aadj[v * NN + w];
        const float scale = 0.95f / rs;        // (1-alpha)/rowsum
        for (int c = tid; c < CC; c += 256) {
            float acc = 0.f;
            for (int w = 0; w < NN; ++w) {
                float aw = Aadj[v * NN + w] + (w == v ? 1.f : 0.f);
                acc += aw * centers[w * CC + c];
            }
            h1[v * CC + c] = 0.05f * centers[v * CC + c] + scale * acc;
        }
    } else if (tid < NN) {
        float s = b_bc[0];
        for (int c = 0; c < CC; ++c) s += W_bc[c] * centers[tid * CC + c];
        stab[tid] = s;
    }
}

// ---------------------------------------------------------------------------
// kB: new_centers[v,o] = b_gcn[o] + W_gcn[o,0:512].centers[v] + W_gcn[o,512:].h1[v]
// ---------------------------------------------------------------------------
__global__ void graphcam_kB(const float* __restrict__ centers,
                            const float* __restrict__ h1,
                            const float* __restrict__ W_gcn,
                            const float* __restrict__ b_gcn,
                            float* __restrict__ newc) {
    const int v = blockIdx.x;
    const int tid = threadIdx.x;
    for (int o = tid; o < CC; o += 256) {
        float acc = b_gcn[o];
        const float* w0 = W_gcn + (size_t)o * (2 * CC);
        const float* cv = centers + v * CC;
        const float* hv = h1 + v * CC;
        for (int c = 0; c < CC; ++c) acc += w0[c] * cv[c];
        for (int c = 0; c < CC; ++c) acc += w0[CC + c] * hv[c];
        newc[v * CC + o] = acc;
    }
}

// ---------------------------------------------------------------------------
// kC: the WMMA kernel. grid = B*72 blocks, 256 threads (8 waves).
//     Block owns 128 contiguous hw positions of image b. Each wave computes a
//     16(M)x64(N) tile, pass 0 vs centers, pass 1 vs new_centers.
// ---------------------------------------------------------------------------
__global__ __launch_bounds__(256) void graphcam_kC(
        const float* __restrict__ x,
        const float* __restrict__ centers,
        const float* __restrict__ gum,
        const float* __restrict__ newc,
        const float* __restrict__ stab,
        float* __restrict__ out,
        float* __restrict__ heatP,
        float* __restrict__ probP) {
    __shared__ unsigned short sCen[NN * CC];      // codebook in bf16 (64 KB)
    __shared__ float sTile[128 * TSTR];           // transposed result tile
    __shared__ float sTab[NN];
    __shared__ float sRv[256];
    __shared__ int   sRn[256];

    const int tid    = threadIdx.x;
    const int lane   = tid & 31;
    const int wave   = tid >> 5;
    const int b      = blockIdx.x / 72;
    const int hwbase = (blockIdx.x % 72) * 128;
    const int m      = lane & 15;          // position within the wave M-tile
    const int lh     = lane >> 4;          // lane half (K offset select)
    const int hw0    = hwbase + wave * 16;

    const int pos    = tid & 127;          // epilogue: position owned
    const int nhalf  = tid >> 7;           // epilogue: n-range half
    const int hw     = hwbase + pos;

    if (tid < NN) sTab[tid] = stab[tid];

    const size_t xbase = ((size_t)b * CC) * HW + hw0 + m;

    for (int pass = 0; pass < 2; ++pass) {
        // ---- stage codebook (bf16) into LDS ----
        const float* src = (pass == 0) ? centers : newc;
        __syncthreads();
        for (int i = tid; i < NN * CC; i += 256) sCen[i] = f2bf(src[i]);
        __syncthreads();

        v8f zero = {0.f, 0.f, 0.f, 0.f, 0.f, 0.f, 0.f, 0.f};
        v8f acc[4];
        acc[0] = zero; acc[1] = zero; acc[2] = zero; acc[3] = zero;

        // ---- K loop: 512 channels, 32 per WMMA ----
        // unroll 4: enough load/WMMA overlap for pipelining, but keeps the
        // wave under 256 VGPRs (no s_set_vgpr_msb churn, better occupancy).
#pragma unroll 4
        for (int k0 = 0; k0 < CC; k0 += 32) {
            // A fragment: 16x32 bf16 (K contiguous: lane<16 -> K=k0+0..15,
            // lane>=16 -> +16), built from strided f32 global loads of x.
            // Native f32->bf16 converts keep inner-loop VALU low.
            AFrag a;
            const float* xp = x + xbase + (size_t)(k0 + lh * 16) * HW;
            float f[16];
#pragma unroll
            for (int e = 0; e < 8; ++e) f[e]     = xp[(size_t)e * HW];
#pragma unroll
            for (int e = 0; e < 8; ++e) f[8 + e] = xp[(size_t)(8 + e) * HW];
#pragma unroll
            for (int e = 0; e < 16; ++e) a.v[e] = (__bf16)f[e];

            // B fragments: 32x16 bf16, lane = column n, K contiguous per lane
            // (lane<16 -> K=k0..k0+15, lane>=16 -> k0+16..k0+31): 32B LDS read.
            // All four fragments loaded first so the scheduler overlaps WMMA
            // issue with outstanding ds_loads (partial s_wait_dscnt).
            const int kb = k0 + lh * 16;
            BFrag bf[4];
#pragma unroll
            for (int nt = 0; nt < 4; ++nt) {
                const uint4* p = (const uint4*)(&sCen[(nt * 16 + m) * CC + kb]);
                bf[nt].q[0] = p[0];
                bf[nt].q[1] = p[1];
            }
#pragma unroll
            for (int nt = 0; nt < 4; ++nt)
                acc[nt] = __builtin_amdgcn_wmma_f32_16x16x32_bf16(
                    false, a.v, false, bf[nt].v, (short)0, acc[nt], false, false);
        }

        // ---- transpose accumulators through padded LDS tile ----
        // C/D layout: VGPR r, lanes<16: M=r,N=lane ; lanes>=16: M=r+8,N=lane-16
#pragma unroll
        for (int nt = 0; nt < 4; ++nt)
#pragma unroll
            for (int r = 0; r < 8; ++r)
                sTile[(wave * 16 + r + 8 * lh) * TSTR + nt * 16 + m] = acc[nt][r];
        __syncthreads();

        if (pass == 0) {
            // inner_products output + gumbel argmax + heatmap
            float bv = -3.4e38f; int bn = 0;
            const float* gb  = gum + ((size_t)b * NN) * HW + hw;
            float*       ipb = out + OFF_IP + ((size_t)b * NN) * HW + hw;
#pragma unroll
            for (int j = 0; j < 32; ++j) {
                int n = nhalf * 32 + j;
                float v = sTile[pos * TSTR + n];
                __builtin_nontemporal_store(v, ipb + (size_t)n * HW);   // stream
                float t = v + __builtin_nontemporal_load(gb + (size_t)n * HW);
                if (t > bv) { bv = t; bn = n; }
            }
            sRv[tid] = bv; sRn[tid] = bn;
            __syncthreads();
            if (nhalf == 0) {
                float v1 = sRv[tid + 128]; int n1 = sRn[tid + 128];
                if (v1 > bv) { bv = v1; bn = n1; }             // tie -> lower n
                float hval = sTab[bn];
                out[(size_t)b * HW + hw] = hval;               // heatmap
                sRv[tid] = hval;
            }
            __syncthreads();
            for (int s = 64; s >= 1; s >>= 1) {
                if (tid < s) sRv[tid] += sRv[tid + s];
                __syncthreads();
            }
            if (tid == 0) heatP[blockIdx.x] = sRv[0];
            __syncthreads();
        } else {
            // max-softmax prob: prob = 1 / sum_n exp(nip[n] - max)
            float vals[32];
            float lmax = -3.4e38f;
#pragma unroll
            for (int j = 0; j < 32; ++j) {
                float v = sTile[pos * TSTR + nhalf * 32 + j];
                vals[j] = v;
                lmax = fmaxf(lmax, v);
            }
            sRv[tid] = lmax;
            __syncthreads();
            float gmax = fmaxf(sRv[pos], sRv[pos + 128]);
            float se = 0.f;
#pragma unroll
            for (int j = 0; j < 32; ++j) se += __expf(vals[j] - gmax);
            __syncthreads();
            sRv[tid] = se;
            __syncthreads();
            if (nhalf == 0) sRv[tid] = 1.0f / (sRv[tid] + sRv[tid + 128]);
            __syncthreads();
            for (int s = 64; s >= 1; s >>= 1) {
                if (tid < s) sRv[tid] += sRv[tid + s];
                __syncthreads();
            }
            if (tid == 0) probP[blockIdx.x] = sRv[0];
        }
    }
}

// ---------------------------------------------------------------------------
// kE: finalize logits (per-image heatmap mean) and cluster_loss.
// ---------------------------------------------------------------------------
__global__ void graphcam_kE(const float* __restrict__ heatP,
                            const float* __restrict__ probP,
                            float* __restrict__ out) {
    const int tid = threadIdx.x;
    if (tid < BB) {
        float s = 0.f;
        for (int j = 0; j < 72; ++j) s += heatP[tid * 72 + j];
        out[OFF_LOGITS + tid] = s * (1.0f / (float)HW);
    } else if (tid == BB) {
        float s = 0.f;
        for (int i = 0; i < BB * 72; ++i) s += probP[i];
        out[OFF_LOSS] = -s * (1.0f / (float)(BB * HW));
    }
}

// ---------------------------------------------------------------------------
extern "C" void kernel_launch(void* const* d_in, const int* in_sizes, int n_in,
                              void* d_out, int out_size, void* d_ws, size_t ws_size,
                              hipStream_t stream) {
    const float* x       = (const float*)d_in[0];
    const float* centers = (const float*)d_in[1];
    const float* Aadj    = (const float*)d_in[2];
    const float* gum     = (const float*)d_in[3];
    const float* W_bc    = (const float*)d_in[4];
    const float* b_bc    = (const float*)d_in[5];
    const float* W_gcn   = (const float*)d_in[6];
    const float* b_gcn   = (const float*)d_in[7];
    float* out = (float*)d_out;
    float* ws  = (float*)d_ws;

    float* h1    = ws;            // 64*512
    float* stab  = ws + 32768;    // 64
    float* newc  = ws + 32832;    // 64*512
    float* heatP = ws + 65600;    // 1152
    float* probP = ws + 66752;    // 1152   (total ~272 KB)

    graphcam_kA<<<65, 256, 0, stream>>>(centers, Aadj, W_bc, b_bc, h1, stab);
    graphcam_kB<<<64, 256, 0, stream>>>(centers, h1, W_gcn, b_gcn, newc);
    graphcam_kC<<<BB * 72, 256, 0, stream>>>(x, centers, gum, newc, stab, out,
                                             heatP, probP);
    graphcam_kE<<<1, 64, 0, stream>>>(heatP, probP, out);
}
